// UPPT_Decoder_66228395704782
// MI455X (gfx1250) — compile-verified, matrix operand order checked by reading, not run on hardware
//
#include <hip/hip_runtime.h>
#include <hip/hip_bf16.h>
#include <math.h>

typedef _Float16 h16;
typedef __attribute__((ext_vector_type(16))) _Float16 v16h;
typedef __attribute__((ext_vector_type(8)))  _Float16 v8h;
typedef __attribute__((ext_vector_type(8)))  float    v8f;

#define IDIM 80
#define EDIM 256
#define ADIM 128
#define GDIM 256
#define BSZ  32
#define TENC 800
#define TDEC 400
#define KIH  352   // 80+256=336 padded to multiple of 32
#define KCAT 512   // GDIM + EDIM

__device__ __forceinline__ v8f wmma_f16(v16h a, v16h b, v8f c) {
  // D = A(16x32 f16) * B(32x16 f16) + C(16x16 f32)
  return __builtin_amdgcn_wmma_f32_16x16x32_f16(false, a, false, b, (short)0, c,
                                                false, false);
}

// A fragment loader: row-major [16][ldk] f16 tile, K offset k0 (multiple of 32).
// lane l: M = l&15, kgrp = (l>>4)*8; elems 0..7 -> K=k0+kgrp+0..7,
// elems 8..15 -> K=k0+kgrp+16..23  (per CDNA5 ISA 16-bit A layout).
__device__ __forceinline__ v16h ldfragA(const h16* base, int ldk, int k0, int lane) {
  int m  = lane & 15;
  int kg = (lane >> 4) << 3;
  const h16* p = base + (size_t)m * ldk + k0 + kg;
  v8h lo = *(const v8h*)p;
  v8h hi = *(const v8h*)(p + 16);
  v16h r;
#pragma unroll
  for (int i = 0; i < 8; ++i) { r[i] = lo[i]; r[i + 8] = hi[i]; }
  return r;
}

// B fragment loader for B[K][N] = W[n][k] (W row-major [Nout][K]).
// lane l: N = n0 + (l&15); same K grouping as A (mirrored layout).
__device__ __forceinline__ v16h ldfragB(const h16* w, int ldk, int n0, int k0, int lane) {
  int nn = n0 + (lane & 15);
  int kg = (lane >> 4) << 3;
  const h16* p = w + (size_t)nn * ldk + k0 + kg;
  v8h lo = *(const v8h*)p;
  v8h hi = *(const v8h*)(p + 16);
  v16h r;
#pragma unroll
  for (int i = 0; i < 8; ++i) { r[i] = lo[i]; r[i + 8] = hi[i]; }
  return r;
}

__device__ __forceinline__ float sigmoid_fast(float x) {
  return 1.f / (1.f + __expf(-x));
}
__device__ __forceinline__ float tanh_fast(float x) {
  return 1.f - 2.f / (__expf(2.f * x) + 1.f);
}

// f32 -> f16 conversion with optional zero column padding (scols -> dcols).
__global__ void cvt_pad_kernel(const float* __restrict__ src, h16* __restrict__ dst,
                               int rows, int scols, int dcols) {
  int n = rows * dcols;
  for (int i = blockIdx.x * blockDim.x + threadIdx.x; i < n;
       i += gridDim.x * blockDim.x) {
    int r = i / dcols, c = i - r * dcols;
    dst[i] = (c < scols) ? (h16)src[(size_t)r * scols + c] : (h16)0.f;
  }
}

// enc_proj = enc_feat @ W_enc^T : (25600 x 256) x (256 x 128) -> f32
// grid = 1600 row tiles; 8 waves per block = 8 column tiles.
__global__ void enc_proj_kernel(const h16* __restrict__ ef,
                                const h16* __restrict__ Wenc,
                                float* __restrict__ ep) {
  const int lane = threadIdx.x & 31, wave = threadIdx.x >> 5;
  const int rt = blockIdx.x;
  const h16* A = ef + (size_t)rt * 16 * EDIM;
  const int n0 = wave * 16;
  v8f acc = {};
#pragma unroll
  for (int k0 = 0; k0 < EDIM; k0 += 32) {
    v16h a = ldfragA(A, EDIM, k0, lane);
    v16h b = ldfragB(Wenc, EDIM, n0, k0, lane);
    acc = wmma_f16(a, b, acc);
  }
  const int nn = lane & 15, mb = (lane >> 4) << 3;
#pragma unroll
  for (int i = 0; i < 8; ++i)
    ep[((size_t)rt * 16 + mb + i) * ADIM + n0 + nn] = acc[i];
}

// Persistent decoder: block bq owns batch rows [bq*16, bq*16+16), 16 waves.
// All inter-phase state (incl. the 16x800 score matrix) lives in LDS:
// gfx1250 WGPs have 320KB LDS, so ~109KB static LDS / workgroup is fine.
__global__ void __launch_bounds__(512, 1)
decoder_kernel(const float* __restrict__ gt,
               const float* __restrict__ b_ih,  const float* __restrict__ b_hh,
               const float* __restrict__ b_attn, const float* __restrict__ v_attn,
               const float* __restrict__ b_out,
               const h16* __restrict__ Wih,  const h16* __restrict__ Whh,
               const h16* __restrict__ Wdec, const h16* __restrict__ Wout,
               const h16* __restrict__ encf_h,   // (32,800,256) f16
               const float* __restrict__ encp,   // (32,800,128) f32
               float* __restrict__ pred_out,     // (32,400,80)
               float* __restrict__ attn_out)     // (32,400,800)
{
  __shared__ __align__(16) h16   inp_h[16 * KIH];    // [x_t | ctx | 0pad] f16
  __shared__ __align__(16) h16   hc_h[16 * KCAT];    // [h | ctx] f16
  __shared__ __align__(16) float h_s[16 * GDIM];     // h f32
  __shared__ __align__(16) float dec_s[16 * ADIM];   // dec_proj + b_attn
  __shared__ __align__(16) float score_s[16 * TENC]; // scores -> attn (51.2KB)
  __shared__ __align__(16) float logit_s[16 * IDIM];
  __shared__ __align__(16) float v_s[ADIM];

  const int tid = threadIdx.x, lane = tid & 31, wave = tid >> 5;
  const int bq = blockIdx.x;
  const int nn = lane & 15, mb = (lane >> 4) << 3;  // D-fragment coords

  for (int i = tid; i < 16 * KIH; i += 512)  inp_h[i] = (h16)0.f;
  for (int i = tid; i < 16 * KCAT; i += 512) hc_h[i]  = (h16)0.f;
  for (int i = tid; i < 16 * GDIM; i += 512) h_s[i]   = 0.f;
  for (int i = tid; i < ADIM; i += 512)      v_s[i]   = v_attn[i];
  __syncthreads();

  for (int t = 0; t < TDEC; ++t) {
    // ---- phase 1: teacher-forced input x_t -> inp_h[:, 0:80] -------------
    for (int i = tid; i < 16 * IDIM; i += 512) {
      int m = i / IDIM, c = i - m * IDIM;
      inp_h[m * KIH + c] =
          (h16)gt[((size_t)(bq * 16 + m) * TDEC + t) * IDIM + c];
    }
    __syncthreads();

    // ---- phase 2: fused GRU (wave w owns gate columns [w*16, w*16+16)) ---
    {
      const int n0 = wave * 16;
      v8f air = {}, aiz = {}, ain = {}, ahr = {}, ahz = {}, ahn = {};
      for (int k0 = 0; k0 < KIH; k0 += 32) {           // gi = inp @ W_ih^T
        v16h a = ldfragA(inp_h, KIH, k0, lane);
        air = wmma_f16(a, ldfragB(Wih, KIH, n0,        k0, lane), air);
        aiz = wmma_f16(a, ldfragB(Wih, KIH, n0 + 256,  k0, lane), aiz);
        ain = wmma_f16(a, ldfragB(Wih, KIH, n0 + 512,  k0, lane), ain);
      }
      for (int k0 = 0; k0 < GDIM; k0 += 32) {          // gh = h @ W_hh^T
        v16h a = ldfragA(hc_h, KCAT, k0, lane);        // cols 0..255 = old h
        ahr = wmma_f16(a, ldfragB(Whh, GDIM, n0,        k0, lane), ahr);
        ahz = wmma_f16(a, ldfragB(Whh, GDIM, n0 + 256,  k0, lane), ahz);
        ahn = wmma_f16(a, ldfragB(Whh, GDIM, n0 + 512,  k0, lane), ahn);
      }
      __syncthreads();  // all waves done reading old h before overwriting it
      const int ncol = n0 + nn;
      const float bir = b_ih[ncol],       bhr = b_hh[ncol];
      const float biz = b_ih[ncol + 256], bhz = b_hh[ncol + 256];
      const float bin = b_ih[ncol + 512], bhn = b_hh[ncol + 512];
#pragma unroll
      for (int i = 0; i < 8; ++i) {
        int m = mb + i;
        float r  = sigmoid_fast(air[i] + bir + ahr[i] + bhr);
        float z  = sigmoid_fast(aiz[i] + biz + ahz[i] + bhz);
        float nv = tanh_fast(ain[i] + bin + r * (ahn[i] + bhn));
        float hn = (1.f - z) * nv + z * h_s[m * GDIM + ncol];
        h_s[m * GDIM + ncol]  = hn;
        hc_h[m * KCAT + ncol] = (h16)hn;
      }
    }
    __syncthreads();

    // ---- phase 3: overlapped GEMMs ---------------------------------------
    // waves 0..7  : dec_proj = h_new @ W_dec^T + b_attn
    // waves 8..12 : h-half (split-K) of pred GEMM, accumulator kept live
    //               across the attention phases; ctx-half finishes in ph.7.
    v8f pacc = {};
    if (wave < 8) {
      const int n0 = wave * 16;
      v8f acc = {};
      for (int k0 = 0; k0 < GDIM; k0 += 32)
        acc = wmma_f16(ldfragA(hc_h, KCAT, k0, lane),
                       ldfragB(Wdec, GDIM, n0, k0, lane), acc);
      const float ba = b_attn[n0 + nn];
#pragma unroll
      for (int i = 0; i < 8; ++i)
        dec_s[(mb + i) * ADIM + n0 + nn] = acc[i] + ba;
    } else if (wave < 13) {
      const int n0 = (wave - 8) * 16;
      for (int k0 = 0; k0 < GDIM; k0 += 32)          // K = 0..255 : h part
        pacc = wmma_f16(ldfragA(hc_h, KCAT, k0, lane),
                        ldfragB(Wout, KCAT, n0, k0, lane), pacc);
    }
    __syncthreads();

    // ---- phase 4: additive attention scores -> score_s (LDS) -------------
#pragma unroll 1
    for (int it = 0; it < 25; ++it) {
      int p  = tid + it * 512;           // 12800 = 16 rows * 800 positions
      int b  = p / TENC, te = p - b * TENC;
      const float* ep = encp + ((size_t)(bq * 16 + b) * TENC + te) * ADIM;
      const float* dp = dec_s + b * ADIM;
      float s = 0.f;
#pragma unroll 4
      for (int k = 0; k < ADIM; k += 4) {
        float4 e = *(const float4*)(ep + k);
        s += tanh_fast(e.x + dp[k + 0]) * v_s[k + 0];
        s += tanh_fast(e.y + dp[k + 1]) * v_s[k + 1];
        s += tanh_fast(e.z + dp[k + 2]) * v_s[k + 2];
        s += tanh_fast(e.w + dp[k + 3]) * v_s[k + 3];
      }
      score_s[b * TENC + te] = s;
    }
    __syncthreads();

    // ---- phase 5: softmax in LDS (wave w -> batch row w); emit attn ------
    {
      float* srow = score_s + wave * TENC;
      float* grow = attn_out + ((size_t)(bq * 16 + wave) * TDEC + t) * TENC;
      float vals[25];
      float mx = -3.4e38f;
#pragma unroll
      for (int i = 0; i < 25; ++i) {
        vals[i] = srow[i * 32 + lane];
        mx = fmaxf(mx, vals[i]);
      }
      for (int o = 16; o; o >>= 1) mx = fmaxf(mx, __shfl_xor(mx, o, 32));
      float sum = 0.f;
#pragma unroll
      for (int i = 0; i < 25; ++i) { vals[i] = __expf(vals[i] - mx); sum += vals[i]; }
      for (int o = 16; o; o >>= 1) sum += __shfl_xor(sum, o, 32);
      const float inv = 1.f / sum;
#pragma unroll
      for (int i = 0; i < 25; ++i) {
        float a = vals[i] * inv;
        srow[i * 32 + lane] = a;      // for ctx reduction (LDS)
        grow[i * 32 + lane] = a;      // attn_record output (global)
      }
    }
    __syncthreads();

    // ---- phase 6: context = attn @ enc_feat (wave w -> batch row w) ------
    {
      const int bg = bq * 16 + wave;
      const float* arow = score_s + wave * TENC;
      const h16* ecol = encf_h + (size_t)bg * TENC * EDIM + lane * 8;
      float acc[8] = {0.f, 0.f, 0.f, 0.f, 0.f, 0.f, 0.f, 0.f};
      for (int te = 0; te < TENC; ++te) {
        __builtin_prefetch(ecol + (size_t)(te + 8) * EDIM, 0, 0);
        float a = arow[te];
        v8h e = *(const v8h*)(ecol + (size_t)te * EDIM);
#pragma unroll
        for (int j = 0; j < 8; ++j) acc[j] += a * (float)e[j];
      }
#pragma unroll
      for (int j = 0; j < 8; ++j) {
        h16 c = (h16)acc[j];
        inp_h[wave * KIH + IDIM + lane * 8 + j] = c;   // ctx for next gi
        hc_h[wave * KCAT + GDIM + lane * 8 + j] = c;   // [h|ctx] for pred
      }
    }
    __syncthreads();

    // ---- phase 7: finish pred GEMM ctx-half (waves 8..12) ----------------
    if (wave >= 8 && wave < 13) {
      const int n0 = (wave - 8) * 16;
      for (int k0 = GDIM; k0 < KCAT; k0 += 32)       // K = 256..511 : ctx part
        pacc = wmma_f16(ldfragA(hc_h, KCAT, k0, lane),
                        ldfragB(Wout, KCAT, n0, k0, lane), pacc);
      const float bo = b_out[n0 + nn];
#pragma unroll
      for (int i = 0; i < 8; ++i)
        logit_s[(mb + i) * IDIM + n0 + nn] = pacc[i] + bo;
    }
    __syncthreads();

    // ---- phase 8: log_softmax rows -> pred_out (wave w -> row w) ---------
    {
      const float* L = logit_s + wave * IDIM;
      float x0 = L[lane];
      float x1 = L[lane + 32];
      float x2 = (lane < 16) ? L[lane + 64] : -3.4e38f;
      float mx = fmaxf(fmaxf(x0, x1), x2);
      for (int o = 16; o; o >>= 1) mx = fmaxf(mx, __shfl_xor(mx, o, 32));
      float sum = __expf(x0 - mx) + __expf(x1 - mx) +
                  ((lane < 16) ? __expf(x2 - mx) : 0.f);
      for (int o = 16; o; o >>= 1) sum += __shfl_xor(sum, o, 32);
      const float lse = mx + __logf(sum);
      float* prow = pred_out + ((size_t)(bq * 16 + wave) * TDEC + t) * IDIM;
      prow[lane]      = x0 - lse;
      prow[lane + 32] = x1 - lse;
      if (lane < 16) prow[lane + 64] = x2 - lse;
    }
    __syncthreads();
  }
}

extern "C" void kernel_launch(void* const* d_in, const int* in_sizes, int n_in,
                              void* d_out, int out_size, void* d_ws, size_t ws_size,
                              hipStream_t stream) {
  const float* enc_feat = (const float*)d_in[0];
  const float* gt       = (const float*)d_in[1];
  const float* W_ih     = (const float*)d_in[2];
  const float* W_hh     = (const float*)d_in[3];
  const float* b_ih     = (const float*)d_in[4];
  const float* b_hh     = (const float*)d_in[5];
  const float* W_enc    = (const float*)d_in[6];
  const float* W_dec    = (const float*)d_in[7];
  const float* b_attn   = (const float*)d_in[8];
  const float* v_attn   = (const float*)d_in[9];
  const float* W_out    = (const float*)d_in[10];
  const float* b_out    = (const float*)d_in[11];

  char* ws = (char*)d_ws;
  size_t off = 0;
  auto alloc = [&](size_t bytes) -> void* {
    off = (off + 255) & ~(size_t)255;
    void* p = ws + off;
    off += bytes;
    return p;
  };
  h16*   Wih_h  = (h16*)alloc((size_t)768 * KIH * 2);
  h16*   Whh_h  = (h16*)alloc((size_t)768 * GDIM * 2);
  h16*   Wdec_h = (h16*)alloc((size_t)ADIM * GDIM * 2);
  h16*   Wout_h = (h16*)alloc((size_t)IDIM * KCAT * 2);
  h16*   Wenc_h = (h16*)alloc((size_t)ADIM * EDIM * 2);
  h16*   encf_h = (h16*)alloc((size_t)BSZ * TENC * EDIM * 2);
  float* encp   = (float*)alloc((size_t)BSZ * TENC * ADIM * 4);

  cvt_pad_kernel<<<(768 * KIH + 255) / 256, 256, 0, stream>>>(W_ih, Wih_h, 768, 336, KIH);
  cvt_pad_kernel<<<(768 * GDIM + 255) / 256, 256, 0, stream>>>(W_hh, Whh_h, 768, GDIM, GDIM);
  cvt_pad_kernel<<<(ADIM * GDIM + 255) / 256, 256, 0, stream>>>(W_dec, Wdec_h, ADIM, GDIM, GDIM);
  cvt_pad_kernel<<<(IDIM * KCAT + 255) / 256, 256, 0, stream>>>(W_out, Wout_h, IDIM, KCAT, KCAT);
  cvt_pad_kernel<<<(ADIM * EDIM + 255) / 256, 256, 0, stream>>>(W_enc, Wenc_h, ADIM, EDIM, EDIM);
  cvt_pad_kernel<<<4096, 256, 0, stream>>>(enc_feat, encf_h, BSZ * TENC, EDIM, EDIM);

  enc_proj_kernel<<<(BSZ * TENC) / 16, 256, 0, stream>>>(encf_h, Wenc_h, encp);

  float* pred_out = (float*)d_out;
  float* attn_out = pred_out + (size_t)BSZ * TDEC * IDIM;
  decoder_kernel<<<2, 512, 0, stream>>>(gt, b_ih, b_hh, b_attn, v_attn, b_out,
                                        Wih_h, Whh_h, Wdec_h, Wout_h,
                                        encf_h, encp, pred_out, attn_out);
}
